// CrossSparseAggrNet_v2_73581379715480
// MI455X (gfx1250) — matrix operand order repaired and checked by reading, not compiled, not run on hardware
//
#include <hip/hip_runtime.h>
#include <hip/hip_bf16.h>
#include <math.h>

// Problem constants (from reference)
#define DIMC 512
#define HID  102
#define KP   39     // KEEPED
#define NKEEP 19    // int(39*0.5)
#define LV   196    // spatial tokens (197-1)
#define LT   64
#define BV   256
#define BT   256
#define NPAD 48     // KP padded to 3 WMMA N-tiles (row 39 = cls, 40..47 = 0)
#define EPSN 1e-12f

typedef __attribute__((ext_vector_type(16))) _Float16 v16h;
typedef __attribute__((ext_vector_type(8)))  _Float16 v8h;
typedef __attribute__((ext_vector_type(8)))  float    v8f;

__device__ inline float wave_sum(float x) {
#pragma unroll
  for (int o = 16; o > 0; o >>= 1) x += __shfl_xor(x, o, 32);
  return x;
}

__device__ inline v16h cat8(v8h lo, v8h hi) {
  return __builtin_shufflevector(lo, hi, 0,1,2,3,4,5,6,7,8,9,10,11,12,13,14,15);
}

// ---------------- K1: layernorm + h = gelu(ln @ W1 + b1) -----------------
// grid (256, 14), block 128. 14 spatial rows per block held in LDS.
__global__ __launch_bounds__(128) void k1_ln_gelu(
    const float* __restrict__ img, const float* __restrict__ g,
    const float* __restrict__ bta, const float* __restrict__ W1,
    const float* __restrict__ b1, float* __restrict__ h) {
  const int b  = blockIdx.x;
  const int l0 = blockIdx.y * 14;
  __shared__ float xs[14][DIMC];
  __shared__ float mu_s[14], rs_s[14];
  const int tid = threadIdx.x;
  for (int idx = tid; idx < 14 * DIMC; idx += 128) {
    int r = idx >> 9, c = idx & 511;
    xs[r][c] = img[((size_t)b * 197 + (l0 + r + 1)) * DIMC + c];
  }
  __syncthreads();
  const int wv = tid >> 5, lane = tid & 31;
  for (int r = wv; r < 14; r += 4) {
    float s = 0.f, q = 0.f;
    for (int j = 0; j < 16; ++j) { float x = xs[r][lane + 32 * j]; s += x; q += x * x; }
    s = wave_sum(s); q = wave_sum(q);
    if (lane == 0) {
      float mu  = s * (1.f / 512.f);
      float var = q * (1.f / 512.f) - mu * mu;
      mu_s[r] = mu; rs_s[r] = rsqrtf(var + 1e-5f);
    }
  }
  __syncthreads();
  for (int idx = tid; idx < 14 * DIMC; idx += 128) {
    int r = idx >> 9, c = idx & 511;
    xs[r][c] = (xs[r][c] - mu_s[r]) * rs_s[r] * g[c] + bta[c];
  }
  __syncthreads();
  if (tid < HID) {
    const int j = tid;
    float acc[14];
    const float bj = b1[j];
#pragma unroll
    for (int r = 0; r < 14; ++r) acc[r] = bj;
    for (int c = 0; c < DIMC; ++c) {
      float w = W1[c * HID + j];   // coalesced across j
#pragma unroll
      for (int r = 0; r < 14; ++r) acc[r] += xs[r][c] * w;  // LDS broadcast
    }
#pragma unroll
    for (int r = 0; r < 14; ++r) {
      float x = acc[r];
      float y = 0.5f * x * (1.f + erff(x * 0.70710678118654752f));  // exact gelu
      h[((size_t)b * LV + (l0 + r)) * HID + j] = y;
    }
  }
}

// ------- K2: w = h@W2+b2 (*scale), softmax over l, aggr = w^T @ spatial -------
// grid 256 (image), block 256.
__global__ __launch_bounds__(256) void k2_softmax_aggr(
    const float* __restrict__ h, const float* __restrict__ img,
    const float* __restrict__ W2, const float* __restrict__ b2,
    const float* __restrict__ scale, float* __restrict__ aggr) {
  const int b = blockIdx.x;
  __shared__ float w2s[HID * KP];
  __shared__ float wm[LV * KP];
  const int tid = threadIdx.x;
  for (int i = tid; i < HID * KP; i += 256) w2s[i] = W2[i];
  __syncthreads();
  const float sc = scale[0];
  for (int e = tid; e < LV * KP; e += 256) {
    int l = e / KP, k = e % KP;
    const float* hr = h + ((size_t)b * LV + l) * HID;
    float s = b2[k];
    for (int j = 0; j < HID; ++j) s += hr[j] * w2s[j * KP + k];
    wm[l * KP + k] = s * sc;
  }
  __syncthreads();
  if (tid < KP) {   // softmax over l per k
    const int k = tid;
    float mx = -3.4e38f;
    for (int l = 0; l < LV; ++l) mx = fmaxf(mx, wm[l * KP + k]);
    float sum = 0.f;
    for (int l = 0; l < LV; ++l) { float e = expf(wm[l * KP + k] - mx); wm[l * KP + k] = e; sum += e; }
    float inv = 1.f / sum;
    for (int l = 0; l < LV; ++l) wm[l * KP + k] *= inv;
  }
  __syncthreads();
  const int c0 = tid, c1 = tid + 256;
  float a0[KP], a1[KP];
#pragma unroll
  for (int k = 0; k < KP; ++k) { a0[k] = 0.f; a1[k] = 0.f; }
  for (int l = 0; l < LV; ++l) {
    const float* sp = img + ((size_t)b * 197 + (l + 1)) * DIMC;
    float s0 = sp[c0], s1 = sp[c1];
#pragma unroll
    for (int k = 0; k < KP; ++k) {
      float w = wm[l * KP + k];            // LDS broadcast
      a0[k] += w * s0; a1[k] += w * s1;
    }
  }
#pragma unroll
  for (int k = 0; k < KP; ++k) {
    aggr[((size_t)b * KP + k) * DIMC + c0] = a0[k];
    aggr[((size_t)b * KP + k) * DIMC + c1] = a1[k];
  }
}

// ---- K3a: per-image prep: rnorm, glo/att_self, f16 Bmat(48x512), Gram ----
__global__ __launch_bounds__(256) void k3a_vprep(
    const float* __restrict__ aggr, const float* __restrict__ img,
    _Float16* __restrict__ Bmat, float* __restrict__ rnorm,
    float* __restrict__ att_self, float* __restrict__ Gram) {
  const int v = blockIdx.x;
  const int tid = threadIdx.x;
  const int wv = tid >> 5, lane = tid & 31;
  __shared__ float rn_s[KP];
  __shared__ float glo_s[DIMC];
  __shared__ float red_s[256];
  const float* ag = aggr + (size_t)v * KP * DIMC;
  for (int k = wv; k < KP; k += 8) {
    float q = 0.f;
    for (int j = 0; j < 16; ++j) { float x = ag[k * DIMC + lane + 32 * j]; q += x * x; }
    q = wave_sum(q);
    if (lane == 0) { float n = fmaxf(sqrtf(q), EPSN); rn_s[k] = n; rnorm[v * KP + k] = n; }
  }
  __syncthreads();
  const int c0 = tid, c1 = tid + 256;
  float m0 = 0.f, m1 = 0.f;
  for (int k = 0; k < KP; ++k) { m0 += ag[k * DIMC + c0]; m1 += ag[k * DIMC + c1]; }
  m0 *= (1.f / KP); m1 *= (1.f / KP);
  red_s[tid] = m0 * m0 + m1 * m1;
  __syncthreads();
  for (int o = 128; o > 0; o >>= 1) { if (tid < o) red_s[tid] += red_s[tid + o]; __syncthreads(); }
  float ginv = 1.f / fmaxf(sqrtf(red_s[0]), EPSN);
  glo_s[c0] = m0 * ginv; glo_s[c1] = m1 * ginv;
  __syncthreads();
  for (int k = wv; k < KP; k += 8) {
    float s = 0.f;
    for (int j = 0; j < 16; ++j) { int c = lane + 32 * j; s += glo_s[c] * ag[k * DIMC + c]; }
    s = wave_sum(s);
    if (lane == 0) att_self[v * KP + k] = s / rn_s[k];
  }
  const float* cls = img + (size_t)v * 197 * DIMC;  // row 0 = img_cls
  float q = cls[c0] * cls[c0] + cls[c1] * cls[c1];
  __syncthreads();
  red_s[tid] = q;
  __syncthreads();
  for (int o = 128; o > 0; o >>= 1) { if (tid < o) red_s[tid] += red_s[tid + o]; __syncthreads(); }
  const float cinv = 1.f / fmaxf(sqrtf(red_s[0]), EPSN);
  _Float16* Bm = Bmat + (size_t)v * NPAD * DIMC;
  for (int i = tid; i < KP * DIMC; i += 256) {
    int k = i >> 9;
    Bm[i] = (_Float16)(ag[i] / rn_s[k]);
  }
  Bm[(size_t)KP * DIMC + c0] = (_Float16)(cls[c0] * cinv);   // row 39 = cls_norm
  Bm[(size_t)KP * DIMC + c1] = (_Float16)(cls[c1] * cinv);
  for (int i = tid; i < 8 * DIMC; i += 256) Bm[(size_t)40 * DIMC + i] = (_Float16)0.f;
  float* G = Gram + (size_t)v * KP * KP;
  for (int e = tid; e < KP * KP; e += 256) {
    int k = e / KP, kk = e % KP;
    float s = 0.f;
    for (int c = 0; c < DIMC; ++c) s += ag[k * DIMC + c] * ag[kk * DIMC + c];
    G[e] = s;
  }
}

// ---- K3b: per-caption prep: f16 cap_norm, masked-mean cap_glo ----
__global__ __launch_bounds__(256) void k3b_tprep(
    const float* __restrict__ cap, const int* __restrict__ lens,
    _Float16* __restrict__ capn, float* __restrict__ cap_glo) {
  const int t = blockIdx.x;
  const int tid = threadIdx.x;
  const int wv = tid >> 5, lane = tid & 31;
  __shared__ float red_s[256];
  const float* cp = cap + (size_t)t * LT * DIMC;
  for (int w = wv; w < LT; w += 8) {
    float q = 0.f;
    for (int j = 0; j < 16; ++j) { float x = cp[w * DIMC + lane + 32 * j]; q += x * x; }
    q = wave_sum(q);                          // xor-reduce: all lanes get full sum
    float inv = 1.f / fmaxf(sqrtf(q), EPSN);
    for (int j = 0; j < 16; ++j) {
      int c = lane + 32 * j;
      capn[((size_t)t * LT + w) * DIMC + c] = (_Float16)(cp[w * DIMC + c] * inv);
    }
  }
  const int len = lens[t];
  const int c0 = tid, c1 = tid + 256;
  float s0 = 0.f, s1 = 0.f;
  for (int w = 0; w < len; ++w) { s0 += cp[w * DIMC + c0]; s1 += cp[w * DIMC + c1]; }
  const float invn = 1.f / (float)len;
  s0 *= invn; s1 *= invn;
  red_s[tid] = s0 * s0 + s1 * s1;
  __syncthreads();
  for (int o = 128; o > 0; o >>= 1) { if (tid < o) red_s[tid] += red_s[tid + o]; __syncthreads(); }
  const float inv = 1.f / fmaxf(sqrtf(red_s[0]), EPSN);
  cap_glo[(size_t)t * DIMC + c0] = s0 * inv;
  cap_glo[(size_t)t * DIMC + c1] = s1 * inv;
}

// ---- K4: per (t,v): score, exact top-19 mask, w_drop, coef via Gram ----
__global__ __launch_bounds__(64) void k4_select(
    const float* __restrict__ aggr, const float* __restrict__ cap_glo,
    const float* __restrict__ att_self, const float* __restrict__ rnorm,
    const float* __restrict__ Gram, unsigned long long* __restrict__ selmask,
    float* __restrict__ coefs) {
  const int t = blockIdx.x, v = blockIdx.y;
  const int p = t * BV + v;
  const int tid = threadIdx.x;
  __shared__ float glo_s[DIMC];
  __shared__ float score[KP];
  __shared__ float wd[KP];
  __shared__ float part[KP];
  __shared__ unsigned long long msk_s;
  __shared__ float smax_s, wsum_s, enorm_s;
  for (int i = tid; i < DIMC; i += 64) glo_s[i] = cap_glo[(size_t)t * DIMC + i];
  __syncthreads();
  if (tid < KP) {
    const float* ar = aggr + ((size_t)v * KP + tid) * DIMC;
    float s = 0.f;
    for (int c = 0; c < DIMC; ++c) s += glo_s[c] * ar[c];
    float att_y = s / rnorm[v * KP + tid];
    score[tid] = 0.8f * att_y + 0.2f * att_self[v * KP + tid];
  }
  __syncthreads();
  if (tid == 0) {
    unsigned long long m = 0ull;
    for (int it = 0; it < NKEEP; ++it) {     // matches top_k tie-breaking (first index)
      float best = -3.4e38f; int bi = 0;
      for (int k = 0; k < KP; ++k)
        if (!((m >> k) & 1ull) && score[k] > best) { best = score[k]; bi = k; }
      m |= (1ull << bi);
    }
    msk_s = m; selmask[p] = m;
    float mx = -3.4e38f;
    for (int k = 0; k < KP; ++k) if (!((m >> k) & 1ull)) mx = fmaxf(mx, score[k]);
    smax_s = mx;
  }
  __syncthreads();
  if (tid < KP) {   // selected terms underflow to exactly 0 (== exp(score-1e10-max))
    wd[tid] = ((msk_s >> tid) & 1ull) ? 0.f : expf(score[tid] - smax_s);
  }
  __syncthreads();
  if (tid == 0) {
    float s = 0.f;
    for (int k = 0; k < KP; ++k) s += wd[k];
    wsum_s = 1.f / s;
  }
  __syncthreads();
  if (tid < KP) wd[tid] *= wsum_s;
  __syncthreads();
  if (tid < KP) {   // ||extra||^2 = wd^T Gram wd
    const float* G = Gram + ((size_t)v * KP + tid) * KP;
    float s = 0.f;
    for (int k = 0; k < KP; ++k) s += wd[k] * G[k];
    part[tid] = wd[tid] * s;
  }
  __syncthreads();
  if (tid == 0) {
    float e2 = 0.f;
    for (int k = 0; k < KP; ++k) e2 += part[k];
    enorm_s = fmaxf(sqrtf(fmaxf(e2, 0.f)), EPSN);
  }
  __syncthreads();
  if (tid < KP) coefs[(size_t)p * KP + tid] = wd[tid] * rnorm[v * KP + tid] / enorm_s;
}

// ---- K5: WMMA similarity GEMM + masked-max epilogue ----
// grid (256 t, 32), block 128 (4 waves); each block loops 8 images v.
__global__ __launch_bounds__(128) void k5_sim_wmma(
    const _Float16* __restrict__ capn, const _Float16* __restrict__ Bmat,
    const unsigned long long* __restrict__ selmask, const float* __restrict__ coefs,
    const int* __restrict__ lens, float* __restrict__ out) {
  const int t  = blockIdx.x;
  const int v0 = blockIdx.y * 8;
  const int tid = threadIdx.x;
  const int wv = tid >> 5, lane = tid & 31;
  const int half = lane >> 4, lm = lane & 15;
  __shared__ float S[64][NPAD + 1];
  __shared__ float coef_s[KP];
  __shared__ float red_s[64];
  __shared__ unsigned long long msk_s;
  const int len = lens[t];
  // A fragment source: row = word (M = wv*16 + lm), two 8-half chunks per k-step
  const _Float16* Arow = capn + ((size_t)t * LT + (wv << 4) + lm) * DIMC;
  const int klo = half << 4;
  for (int vi = 0; vi < 8; ++vi) {
    const int v = v0 + vi;
    const int p = t * BV + v;
    const _Float16* Bbase = Bmat + (size_t)v * NPAD * DIMC + (half << 4);
    const _Float16* B0 = Bbase + (size_t)(0  + lm) * DIMC;
    const _Float16* B1 = Bbase + (size_t)(16 + lm) * DIMC;
    const _Float16* B2 = Bbase + (size_t)(32 + lm) * DIMC;
    v8f c0 = {}; v8f c1 = {}; v8f c2 = {};
#pragma unroll 4
    for (int s = 0; s < 16; ++s) {
      const int cb = s << 5;   // 32 channels per WMMA K step
      v16h a  = cat8(*(const v8h*)(Arow + cb + klo), *(const v8h*)(Arow + cb + 8 + klo));
      v16h b0 = cat8(*(const v8h*)(B0 + cb), *(const v8h*)(B0 + cb + 8));
      v16h b1 = cat8(*(const v8h*)(B1 + cb), *(const v8h*)(B1 + cb + 8));
      v16h b2 = cat8(*(const v8h*)(B2 + cb), *(const v8h*)(B2 + cb + 8));
      c0 = __builtin_amdgcn_wmma_f32_16x16x32_f16(false, a, false, b0, (short)0, c0, false, false);
      c1 = __builtin_amdgcn_wmma_f32_16x16x32_f16(false, a, false, b1, (short)0, c1, false, false);
      c2 = __builtin_amdgcn_wmma_f32_16x16x32_f16(false, a, false, b2, (short)0, c2, false, false);
    }
#pragma unroll
    for (int r = 0; r < 8; ++r) {   // C/D layout: M = r + 8*half
      const int row = (wv << 4) + r + (half << 3);
      S[row][0  + lm] = c0[r];
      S[row][16 + lm] = c1[r];
      S[row][32 + lm] = c2[r];
    }
    if (tid < KP) coef_s[tid] = coefs[(size_t)p * KP + tid];
    if (tid == 0) msk_s = selmask[p];
    __syncthreads();
    if (tid < 64) {
      const int w = tid;
      const unsigned long long m = msk_s;
      float mx = -3.4e38f, p1 = 0.f;
#pragma unroll
      for (int k = 0; k < KP; ++k) {
        float sv = S[w][k];
        if ((m >> k) & 1ull) mx = fmaxf(mx, sv);  // selected aggr rows
        p1 += coef_s[k] * sv;                     // extra_norm dot, reconstructed
      }
      mx = fmaxf(mx, S[w][KP]);                   // cls row (n = 39)
      mx = fmaxf(mx, p1);
      red_s[w] = (w < len) ? mx : 0.f;
    }
    __syncthreads();
    if (tid == 0) {
      float s = 0.f;
      for (int w = 0; w < 64; ++w) s += red_s[w];
      out[(size_t)v * BT + t] = s / (float)len;   // sims.T layout
    }
    __syncthreads();
  }
}

extern "C" void kernel_launch(void* const* d_in, const int* in_sizes, int n_in,
                              void* d_out, int out_size, void* d_ws, size_t ws_size,
                              hipStream_t stream) {
  (void)in_sizes; (void)n_in; (void)out_size; (void)ws_size;
  const float* img   = (const float*)d_in[0];
  const float* cap   = (const float*)d_in[1];
  const int*   lens  = (const int*)d_in[2];
  const float* ln_g  = (const float*)d_in[3];
  const float* ln_b  = (const float*)d_in[4];
  const float* W1    = (const float*)d_in[5];
  const float* b1    = (const float*)d_in[6];
  const float* W2    = (const float*)d_in[7];
  const float* b2    = (const float*)d_in[8];
  const float* scale = (const float*)d_in[9];
  float* out = (float*)d_out;

  char* ws = (char*)d_ws;
  size_t off = 0;
  auto carve = [&](size_t bytes) -> void* {
    off = (off + 255) & ~(size_t)255;
    void* p = ws + off;
    off += bytes;
    return p;
  };
  float*    h        = (float*)carve((size_t)BV * LV * HID * 4);      // 20.5 MB
  float*    aggr     = (float*)carve((size_t)BV * KP * DIMC * 4);     // 20.4 MB
  _Float16* capn     = (_Float16*)carve((size_t)BT * LT * DIMC * 2);  // 16.8 MB
  _Float16* Bmat     = (_Float16*)carve((size_t)BV * NPAD * DIMC * 2);// 12.6 MB
  float*    cap_glo  = (float*)carve((size_t)BT * DIMC * 4);
  float*    rnorm    = (float*)carve((size_t)BV * KP * 4);
  float*    att_self = (float*)carve((size_t)BV * KP * 4);
  float*    Gram     = (float*)carve((size_t)BV * KP * KP * 4);       // 1.6 MB
  unsigned long long* smask = (unsigned long long*)carve((size_t)BT * BV * 8);
  float*    coefs    = (float*)carve((size_t)BT * BV * KP * 4);       // 10.2 MB

  k1_ln_gelu   <<<dim3(BV, 14), 128, 0, stream>>>(img, ln_g, ln_b, W1, b1, h);
  k2_softmax_aggr<<<dim3(BV),   256, 0, stream>>>(h, img, W2, b2, scale, aggr);
  k3a_vprep    <<<dim3(BV),     256, 0, stream>>>(aggr, img, Bmat, rnorm, att_self, Gram);
  k3b_tprep    <<<dim3(BT),     256, 0, stream>>>(cap, lens, capn, cap_glo);
  k4_select    <<<dim3(BT, BV),  64, 0, stream>>>(aggr, cap_glo, att_self, rnorm, Gram, smask, coefs);
  k5_sim_wmma  <<<dim3(BT, 32), 128, 0, stream>>>(capn, Bmat, smask, coefs, lens, out);
}
